// DummyGPt_67748814127593
// MI455X (gfx1250) — compile-verified, matrix operand order checked by reading, not run on hardware
//
#include <hip/hip_runtime.h>
#include <hip/hip_bf16.h>
#include <math.h>

typedef _Float16 h16;
typedef _Float16 v16h __attribute__((ext_vector_type(16)));
typedef _Float16 v8h  __attribute__((ext_vector_type(8)));
typedef float    v8f  __attribute__((ext_vector_type(8)));

constexpr int Bc = 2, Sc = 512, Dc = 768, Hc = 12, NLc = 6, Vc = 32000;
constexpr int HDc = 64, Fc = 3072, Mc = Bc * Sc;   // 1024 token rows
constexpr float EPSc = 1e-6f;

#if __has_builtin(__builtin_amdgcn_global_load_async_to_lds_b128) && \
    __has_builtin(__builtin_amdgcn_s_wait_asynccnt)
#define HAVE_ASYNC_LDS 1
#else
#define HAVE_ASYNC_LDS 0
#endif

#if HAVE_ASYNC_LDS
// Builtin signature (per hipcc diagnostic): params are pointers to
// int __attribute__((vector_size(16))) in AS1 (global) / AS3 (LDS).
typedef int gv4i __attribute__((vector_size(16)));
typedef __attribute__((address_space(1))) gv4i* as1_v4i;
typedef __attribute__((address_space(3))) gv4i* as3_v4i;

// 16-byte async global->LDS copy (per-lane addresses), ASYNCcnt-tracked.
__device__ inline void async_cp16(const h16* g, h16* l) {
  __builtin_amdgcn_global_load_async_to_lds_b128(
      (as1_v4i)(uintptr_t)g,
      (as3_v4i)(uint32_t)(uintptr_t)l,
      0, 0);
}
#endif

// ---------------------------------------------------------------------------
// WMMA helpers. Per ISA 7.12.2 (16-bit A 16x32): lane L holds row (A) /
// col (B) = L%16; halves j in [0,8) are K = 8*(L/16)+j, halves j in [8,16)
// are K = 16 + 8*(L/16) + (j-8)  -> two contiguous 16-byte loads per lane.
// ---------------------------------------------------------------------------
__device__ inline v16h load_tile(const h16* __restrict__ p0, int stride, int lane) {
  const int g = lane >> 4;
  const int r = lane & 15;
  const h16* p = p0 + (size_t)r * stride + g * 8;
  v8h lo = *(const v8h*)(p);
  v8h hi = *(const v8h*)(p + 16);
  v16h o;
#pragma unroll
  for (int i = 0; i < 8; ++i) { o[i] = lo[i]; o[i + 8] = hi[i]; }
  return o;
}

__device__ inline v8f wmma32(v16h a, v16h b, v8f c) {
  return __builtin_amdgcn_wmma_f32_16x16x32_f16(false, a, false, b, (short)0, c,
                                                false, false);
}

__device__ inline float gelu_f(float t) {
  float u = 0.7978845608028654f * (t + 0.044715f * t * t * t);
  return 0.5f * t * (1.0f + tanhf(u));
}

// ---------------------------------------------------------------------------
// Embedding: X[m,d] = token_emb[x[m]][d] + pos_emb[m%S][d]   (f32)
// ---------------------------------------------------------------------------
__global__ void embed_k(const int* __restrict__ x, const float* __restrict__ tok,
                        const float* __restrict__ pos, float* __restrict__ X) {
  int idx = blockIdx.x * 256 + threadIdx.x;
  if (idx >= Mc * Dc) return;
  int m = idx / Dc, d = idx - m * Dc;
  int s = m & (Sc - 1);
  int t = x[m];
  X[idx] = tok[(size_t)t * Dc + d] + pos[(size_t)s * Dc + d];
}

// ---------------------------------------------------------------------------
// f32 [K,N] -> f16 transposed [N,K], LDS-tiled 32x32 so both the f32 reads
// and the f16 writes are coalesced.  grid = (N/32, K/32), block = (32,8).
// ---------------------------------------------------------------------------
__global__ void conv_t_k(const float* __restrict__ in, h16* __restrict__ out,
                         int K, int N) {
  __shared__ float tile[32][33];
  const int n0 = blockIdx.x * 32, k0 = blockIdx.y * 32;
  const int tx = threadIdx.x, ty = threadIdx.y;
#pragma unroll
  for (int r = 0; r < 4; ++r)
    tile[ty + 8 * r][tx] = in[(size_t)(k0 + ty + 8 * r) * N + n0 + tx];
  __syncthreads();
#pragma unroll
  for (int r = 0; r < 4; ++r)
    out[(size_t)(n0 + ty + 8 * r) * K + k0 + tx] = (h16)tile[tx][ty + 8 * r];
}

// V16 [M,D] -> VT [B,H,HD,S]  (per-head transpose so P@V uses B^T rows)
__global__ void transpose_v_k(const h16* __restrict__ V16, h16* __restrict__ VT) {
  int idx = blockIdx.x * 256 + threadIdx.x;
  if (idx >= Mc * Dc) return;
  int m = idx / Dc, df = idx - m * Dc;
  int b = m / Sc, s = m & (Sc - 1);
  int h = df >> 6, d = df & 63;
  VT[(((size_t)(b * Hc + h) * HDc + d) * Sc) + s] = V16[idx];
}

// ---------------------------------------------------------------------------
// DummyNorm: y = s*(x-mean)/(std_ddof1 + eps) + b  -> f16 output row
// ---------------------------------------------------------------------------
__global__ void norm_k(const float* __restrict__ X, const float* __restrict__ sc,
                       const float* __restrict__ bi, h16* __restrict__ out) {
  __shared__ float red[256];
  const int row = blockIdx.x, t = threadIdx.x;
  const float* xp = X + (size_t)row * Dc;
  float x0 = xp[t], x1 = xp[t + 256], x2 = xp[t + 512];
  red[t] = x0 + x1 + x2;
  __syncthreads();
  for (int s = 128; s > 0; s >>= 1) {
    if (t < s) red[t] += red[t + s];
    __syncthreads();
  }
  float mean = red[0] * (1.0f / Dc);
  __syncthreads();
  float d0 = x0 - mean, d1 = x1 - mean, d2 = x2 - mean;
  red[t] = d0 * d0 + d1 * d1 + d2 * d2;
  __syncthreads();
  for (int s = 128; s > 0; s >>= 1) {
    if (t < s) red[t] += red[t + s];
    __syncthreads();
  }
  float var = red[0] * (1.0f / (Dc - 1));   // ddof=1
  float inv = 1.0f / (sqrtf(var) + EPSc);
  h16* op = out + (size_t)row * Dc;
  op[t]       = (h16)(sc[t]       * d0 * inv + bi[t]);
  op[t + 256] = (h16)(sc[t + 256] * d1 * inv + bi[t + 256]);
  op[t + 512] = (h16)(sc[t + 512] * d2 * inv + bi[t + 512]);
}

// ---------------------------------------------------------------------------
// Generic GEMM with LDS-staged B: C = A[M,K](f16) * B (via B^T rows [N,K] f16)
// Block = 256 thr (8 waves) -> 128x64 out tile.  Per K-step the block stages
// the shared 64x32 B^T tile (4 KB) into LDS in per-lane WMMA order (one
// 16-byte copy per thread): 8x less global/L2 B traffic than per-wave loads.
// With the gfx1250 async-to-LDS path the stage is double-buffered: the DMA of
// tile k+1 overlaps the WMMAs on tile k, one barrier per K-step, ASYNCcnt
// synchronized (each thread waits its own copy before the barrier).
// MODE 0: out f16; 1: f16 gelu(acc+bias); 2: fout += acc+bias; 3: fout = acc+bias
// ---------------------------------------------------------------------------
template <int MODE>
__global__ void gemm_k(const h16* __restrict__ A, const h16* __restrict__ BT,
                       const float* __restrict__ bias, float* __restrict__ fout,
                       h16* __restrict__ hout, int Mn, int Nn, int Kn) {
  __shared__ alignas(32) h16 Bs[2][4 * 32 * 16];  // 2 x (4 n-tiles * 512 h16)
  const int tid = threadIdx.x;
  const int lane = tid & 31;
  const int wave = tid >> 5;
  const int m0 = blockIdx.y * 128 + wave * 16;
  const int n0 = blockIdx.x * 64;

  // cooperative B-stage mapping: tid -> (tile ci, lane cl, 16B-chunk cj)
  const int ci = tid >> 6;
  const int cl = (tid >> 1) & 31;
  const int cj = tid & 1;
  const h16* bsrc = BT + (size_t)(n0 + 16 * ci + (cl & 15)) * Kn +
                    (cj ? 16 : 0) + 8 * (cl >> 4);
  const int boff = (ci * 32 + cl) * 16 + cj * 8;

  v8f acc[4] = {};
  const h16* Ap = A + (size_t)m0 * Kn;

#if HAVE_ASYNC_LDS
  async_cp16(bsrc, &Bs[0][boff]);               // prologue: stage k0 = 0
  int buf = 0;
  for (int k0 = 0; k0 < Kn; k0 += 32) {
    __builtin_amdgcn_s_wait_asynccnt(0);        // own chunk landed
    __syncthreads();                            // everyone's chunk landed
    if (k0 + 32 < Kn)                           // overlap next stage w/ WMMA
      async_cp16(bsrc + k0 + 32, &Bs[buf ^ 1][boff]);
    v16h a = load_tile(Ap + k0, Kn, lane);
#pragma unroll
    for (int i = 0; i < 4; ++i) {
      v16h b = *(const v16h*)&Bs[buf][(i * 32 + lane) * 16];
      acc[i] = wmma32(a, b, acc[i]);
    }
    buf ^= 1;
  }
#else
  for (int k0 = 0; k0 < Kn; k0 += 32) {
    __syncthreads();                            // previous-iteration reads done
    *(v8h*)&Bs[0][boff] = *(const v8h*)(bsrc + k0);
    __syncthreads();                            // stage visible to all waves
    v16h a = load_tile(Ap + k0, Kn, lane);
#pragma unroll
    for (int i = 0; i < 4; ++i) {
      v16h b = *(const v16h*)&Bs[0][(i * 32 + lane) * 16];
      acc[i] = wmma32(a, b, acc[i]);
    }
  }
#endif

  const int g = lane >> 4, cc = lane & 15;
#pragma unroll
  for (int i = 0; i < 4; ++i) {
    int n = n0 + 16 * i + cc;
    float bv = (MODE != 0) ? bias[n] : 0.0f;
#pragma unroll
    for (int r = 0; r < 8; ++r) {
      int m = m0 + r + 8 * g;
      size_t off = (size_t)m * Nn + n;
      float t = acc[i][r];
      if (MODE == 0)      hout[off] = (h16)t;
      else if (MODE == 1) hout[off] = (h16)gelu_f(t + bv);
      else if (MODE == 2) fout[off] += t + bv;
      else                fout[off] = t + bv;
    }
  }
}

// ---------------------------------------------------------------------------
// scores[bh,q,k] = scale * dot(Q[b,q,h,:], K[b,k,h,:])   (K-dim = 64 -> 2 WMMAs)
// ---------------------------------------------------------------------------
__global__ void attn_scores_k(const h16* __restrict__ Q, const h16* __restrict__ Kmat,
                              float* __restrict__ SC, float scale) {
  const int lane = threadIdx.x & 31, wave = threadIdx.x >> 5;
  const int bh = blockIdx.y;
  const int b = bh / Hc, h = bh - b * Hc;
  const h16* Qp = Q + (size_t)b * Sc * Dc + h * HDc;
  const h16* Kp = Kmat + (size_t)b * Sc * Dc + h * HDc;
  const int m0 = blockIdx.x * 16;
  float* SCp = SC + (size_t)bh * Sc * Sc;
  v16h a0 = load_tile(Qp + (size_t)m0 * Dc, Dc, lane);
  v16h a1 = load_tile(Qp + (size_t)m0 * Dc + 32, Dc, lane);
  const int g = lane >> 4, cc = lane & 15;
  for (int nt = wave; nt < Sc / 16; nt += 8) {
    int n0 = nt * 16;
    v8f acc = {};
    v16h b0 = load_tile(Kp + (size_t)n0 * Dc, Dc, lane);
    acc = wmma32(a0, b0, acc);
    v16h b1 = load_tile(Kp + (size_t)n0 * Dc + 32, Dc, lane);
    acc = wmma32(a1, b1, acc);
#pragma unroll
    for (int r = 0; r < 8; ++r)
      SCp[(size_t)(m0 + r + 8 * g) * Sc + n0 + cc] = acc[r] * scale;
  }
}

// Softmax over keys with reference's mask: key k survives iff k >= q.
__global__ void softmax_k(const float* __restrict__ SC, h16* __restrict__ P) {
  __shared__ float red[256];
  const int row = blockIdx.x;           // bh*S + q
  const int q = row & (Sc - 1);
  const float* src = SC + (size_t)row * Sc;
  h16* dst = P + (size_t)row * Sc;
  const int t = threadIdx.x;
  const int k0 = t, k1 = t + 256;
  float v0 = (k0 >= q) ? src[k0] : -INFINITY;
  float v1 = (k1 >= q) ? src[k1] : -INFINITY;
  red[t] = fmaxf(v0, v1);
  __syncthreads();
  for (int s = 128; s > 0; s >>= 1) {
    if (t < s) red[t] = fmaxf(red[t], red[t + s]);
    __syncthreads();
  }
  float mx = red[0];
  __syncthreads();
  float e0 = (k0 >= q) ? expf(v0 - mx) : 0.0f;
  float e1 = (k1 >= q) ? expf(v1 - mx) : 0.0f;
  red[t] = e0 + e1;
  __syncthreads();
  for (int s = 128; s > 0; s >>= 1) {
    if (t < s) red[t] += red[t + s];
    __syncthreads();
  }
  float inv = 1.0f / red[0];
  dst[k0] = (h16)(e0 * inv);
  dst[k1] = (h16)(e1 * inv);
}

// ctx[b,q,h,d] = sum_k P[bh,q,k] * V[b,k,h,d]; uses VT rows, K-dim = 512.
__global__ void attn_ctx_k(const h16* __restrict__ P, const h16* __restrict__ VT,
                           h16* __restrict__ CTX) {
  const int lane = threadIdx.x & 31, wave = threadIdx.x >> 5;
  const int bh = blockIdx.y;
  const int b = bh / Hc, h = bh - b * Hc;
  const h16* Pp = P + (size_t)bh * Sc * Sc;
  const h16* Vp = VT + (size_t)bh * HDc * Sc;
  const int m0 = blockIdx.x * 128 + wave * 16;
  v8f acc[4] = {};
  for (int k0 = 0; k0 < Sc; k0 += 32) {
    v16h a = load_tile(Pp + (size_t)m0 * Sc + k0, Sc, lane);
#pragma unroll
    for (int i = 0; i < 4; ++i) {
      v16h bb = load_tile(Vp + (size_t)(16 * i) * Sc + k0, Sc, lane);
      acc[i] = wmma32(a, bb, acc[i]);
    }
  }
  const int g = lane >> 4, cc = lane & 15;
#pragma unroll
  for (int i = 0; i < 4; ++i)
#pragma unroll
    for (int r = 0; r < 8; ++r)
      CTX[(size_t)(b * Sc + m0 + r + 8 * g) * Dc + h * HDc + 16 * i + cc] =
          (h16)acc[i][r];
}

// ---------------------------------------------------------------------------
extern "C" void kernel_launch(void* const* d_in, const int* in_sizes, int n_in,
                              void* d_out, int out_size, void* d_ws, size_t ws_size,
                              hipStream_t stream) {
  (void)in_sizes; (void)n_in; (void)out_size; (void)ws_size;
  const int*   xin   = (const int*)d_in[0];
  const float* tokE  = (const float*)d_in[1];
  const float* posE  = (const float*)d_in[2];
  const float* n1s   = (const float*)d_in[3];
  const float* n1b   = (const float*)d_in[4];
  const float* n2s   = (const float*)d_in[5];
  const float* n2b   = (const float*)d_in[6];
  const float* wq    = (const float*)d_in[7];
  const float* wk    = (const float*)d_in[8];
  const float* wv    = (const float*)d_in[9];
  const float* wo    = (const float*)d_in[10];
  const float* bo    = (const float*)d_in[11];
  const float* w1    = (const float*)d_in[12];
  const float* b1    = (const float*)d_in[13];
  const float* w2    = (const float*)d_in[14];
  const float* b2    = (const float*)d_in[15];
  const float* fsc   = (const float*)d_in[16];
  const float* fbi   = (const float*)d_in[17];
  const float* headw = (const float*)d_in[18];
  const float* headb = (const float*)d_in[19];
  float* OUT = (float*)d_out;

  char* ws = (char*)d_ws;
  size_t off = 0;
  auto alloc = [&](size_t bytes) -> void* {
    void* p = ws + off;
    off = (off + bytes + 255) & ~(size_t)255;
    return p;
  };
  float* X     = (float*)alloc((size_t)Mc * Dc * 4);
  h16*  HN16   = (h16*)alloc((size_t)Mc * Dc * 2);
  h16*  Q16    = (h16*)alloc((size_t)Mc * Dc * 2);
  h16*  K16    = (h16*)alloc((size_t)Mc * Dc * 2);
  h16*  V16    = (h16*)alloc((size_t)Mc * Dc * 2);
  h16*  VT16   = (h16*)alloc((size_t)Mc * Dc * 2);
  h16*  CTX16  = (h16*)alloc((size_t)Mc * Dc * 2);
  float* SC32  = (float*)alloc((size_t)Bc * Hc * Sc * Sc * 4);
  h16*  P16    = (h16*)alloc((size_t)Bc * Hc * Sc * Sc * 2);
  h16*  FF116  = (h16*)alloc((size_t)Mc * Fc * 2);
  h16*  WQT    = (h16*)alloc((size_t)Dc * Dc * 2);
  h16*  WKT    = (h16*)alloc((size_t)Dc * Dc * 2);
  h16*  WVT    = (h16*)alloc((size_t)Dc * Dc * 2);
  h16*  WOT    = (h16*)alloc((size_t)Dc * Dc * 2);
  h16*  W1T    = (h16*)alloc((size_t)Dc * Fc * 2);
  h16*  W2T    = (h16*)alloc((size_t)Fc * Dc * 2);
  h16*  HWT    = (h16*)alloc((size_t)Dc * Vc * 2);

  const dim3 blk(256);
  const dim3 tblk(32, 8);
  embed_k<<<(Mc * Dc + 255) / 256, blk, 0, stream>>>(xin, tokE, posE, X);
  conv_t_k<<<dim3(Vc / 32, Dc / 32), tblk, 0, stream>>>(headw, HWT, Dc, Vc);

  for (int l = 0; l < NLc; ++l) {
    const size_t odd = (size_t)l * Dc * Dc;
    const size_t odf = (size_t)l * Dc * Fc;
    conv_t_k<<<dim3(Dc / 32, Dc / 32), tblk, 0, stream>>>(wq + odd, WQT, Dc, Dc);
    conv_t_k<<<dim3(Dc / 32, Dc / 32), tblk, 0, stream>>>(wk + odd, WKT, Dc, Dc);
    conv_t_k<<<dim3(Dc / 32, Dc / 32), tblk, 0, stream>>>(wv + odd, WVT, Dc, Dc);
    conv_t_k<<<dim3(Dc / 32, Dc / 32), tblk, 0, stream>>>(wo + odd, WOT, Dc, Dc);
    conv_t_k<<<dim3(Fc / 32, Dc / 32), tblk, 0, stream>>>(w1 + odf, W1T, Dc, Fc);
    conv_t_k<<<dim3(Dc / 32, Fc / 32), tblk, 0, stream>>>(w2 + odf, W2T, Fc, Dc);

    norm_k<<<Mc, blk, 0, stream>>>(X, n1s + (size_t)l * Dc, n1b + (size_t)l * Dc, HN16);

    gemm_k<0><<<dim3(Dc / 64, Mc / 128), blk, 0, stream>>>(HN16, WQT, nullptr, nullptr, Q16, Mc, Dc, Dc);
    gemm_k<0><<<dim3(Dc / 64, Mc / 128), blk, 0, stream>>>(HN16, WKT, nullptr, nullptr, K16, Mc, Dc, Dc);
    gemm_k<0><<<dim3(Dc / 64, Mc / 128), blk, 0, stream>>>(HN16, WVT, nullptr, nullptr, V16, Mc, Dc, Dc);
    transpose_v_k<<<(Mc * Dc + 255) / 256, blk, 0, stream>>>(V16, VT16);

    attn_scores_k<<<dim3(Sc / 16, Bc * Hc), blk, 0, stream>>>(Q16, K16, SC32, 0.125f);
    softmax_k<<<Bc * Hc * Sc, blk, 0, stream>>>(SC32, P16);
    attn_ctx_k<<<dim3(Sc / 128, Bc * Hc), blk, 0, stream>>>(P16, VT16, CTX16);

    gemm_k<2><<<dim3(Dc / 64, Mc / 128), blk, 0, stream>>>(CTX16, WOT, bo + (size_t)l * Dc, X, nullptr, Mc, Dc, Dc);

    norm_k<<<Mc, blk, 0, stream>>>(X, n2s + (size_t)l * Dc, n2b + (size_t)l * Dc, HN16);
    gemm_k<1><<<dim3(Fc / 64, Mc / 128), blk, 0, stream>>>(HN16, W1T, b1 + (size_t)l * Fc, nullptr, FF116, Mc, Fc, Dc);
    gemm_k<2><<<dim3(Dc / 64, Mc / 128), blk, 0, stream>>>(FF116, W2T, b2 + (size_t)l * Dc, X, nullptr, Mc, Dc, Fc);
  }

  norm_k<<<Mc, blk, 0, stream>>>(X, fsc, fbi, HN16);
  gemm_k<3><<<dim3(Vc / 64, Mc / 128), blk, 0, stream>>>(HN16, HWT, headb, OUT, nullptr, Mc, Vc, Dc);
}
